// ASTGCN_block_47270410060060
// MI455X (gfx1250) — compile-verified, matrix operand order checked by reading, not run on hardware
//
#include <hip/hip_runtime.h>

// ---------------------------------------------------------------------------
// ASTGCN block for MI455X (gfx1250, wave32).
// All four GEMM-shaped stages run on v_wmma_f32_16x16x32_f16:
//   (1) spatial-attention  S_pre = sig . V_s^T          (B x [320x320x320])
//   (2) xTheta             x_t . Theta_k                (K*B*T x [320x64x64])
//   (3) chebyshev conv     A_k^T . xTheta_k  (+ReLU)    (B x [320x768x960])
//   (4) conv tail          res + temporal conv (+LN)    (B*T x [320x64x{64..192}])
// Every WMMA operand is stored so that the K dimension is contiguous,
// giving 2x b128 loads per fragment per lane.
// ---------------------------------------------------------------------------

constexpr int BB   = 64;            // batch
constexpr int NN   = 307;           // nodes
constexpr int FF   = 64;            // features
constexpr int TT   = 12;            // time steps
constexpr int KK   = 3;             // Chebyshev order
constexpr int CC   = 64;            // gcn channels
constexpr int CTO  = 64;            // output channels
constexpr int NPAD = 320;           // 307 padded to 20 WMMA tiles
constexpr int CTT  = CC * TT;       // 768 fused (c,t) columns
constexpr float LN_EPS = 1e-5f;

typedef __attribute__((ext_vector_type(16))) _Float16 v16h;
typedef __attribute__((ext_vector_type(8)))  _Float16 v8h;
typedef __attribute__((ext_vector_type(8)))  float    v8f;

// Fragment loader for a row-major matrix whose contiguous axis is the WMMA
// K axis.  Works for the A operand (r0 = m0) and for a transposed-B operand
// Bt[n][k] (r0 = n0).  Per CDNA5 ISA 16-bit A layout:
//   lane<16 : row r0+lane,    K = k0+{0..7, 16..23}
//   lane>=16: row r0+lane-16, K = k0+{8..15, 24..31}
// Each half-run is 8 contiguous f16 = one 16-byte load.
__device__ inline v16h load_frag(const _Float16* P, int ld, int r0, int k0, int lane) {
  int row = r0 + (lane & 15);
  int kb  = k0 + ((lane >> 4) << 3);
  const _Float16* p = P + (size_t)row * ld + kb;
  v8h lo = *(const v8h*)(p);
  v8h hi = *(const v8h*)(p + 16);
  return __builtin_shufflevector(lo, hi, 0, 1, 2, 3, 4, 5, 6, 7,
                                 8, 9, 10, 11, 12, 13, 14, 15);
}

// ---------------- 1. temporal attention: E[b,u,t] (softmax over u) ----------
__global__ void k_temporal_attn(const float* __restrict__ x, const float* __restrict__ U1,
                                const float* __restrict__ U2, const float* __restrict__ U3,
                                const float* __restrict__ b_e, const float* __restrict__ V_e,
                                float* __restrict__ E) {
  __shared__ float rhs[NN * TT];     // rhs[n][t] = sum_f U3[f]*x[b,n,f,t]
  __shared__ float aTF[TT * FF];     // a[t][f]   = sum_n x[b,n,f,t]*U1[n]
  __shared__ float lhsTN[TT * NN];   // lhs[t][n'] = a[t,:] . U2[:,n']
  __shared__ float prod[TT * TT];
  __shared__ float eUT[TT * TT];
  const int b = blockIdx.x;
  const int tid = threadIdx.x;
  const float* xb = x + (size_t)b * NN * FF * TT;

  for (int idx = tid; idx < NN * TT; idx += blockDim.x) {
    int n = idx / TT, t = idx % TT;
    float s = 0.f;
    const float* xp = xb + (size_t)n * FF * TT + t;
    for (int f = 0; f < FF; ++f) s += U3[f] * xp[f * TT];
    rhs[idx] = s;
  }
  for (int idx = tid; idx < TT * FF; idx += blockDim.x) {
    int t = idx / FF, f = idx % FF;
    float s = 0.f;
    const float* xp = xb + f * TT + t;
    for (int n = 0; n < NN; ++n) s += xp[(size_t)n * FF * TT] * U1[n];
    aTF[idx] = s;
  }
  __syncthreads();
  for (int idx = tid; idx < TT * NN; idx += blockDim.x) {
    int t = idx / NN, n2 = idx % NN;
    float s = 0.f;
    for (int f = 0; f < FF; ++f) s += aTF[t * FF + f] * U2[f * NN + n2];
    lhsTN[idx] = s;
  }
  __syncthreads();
  for (int idx = tid; idx < TT * TT; idx += blockDim.x) {
    int t = idx / TT, s2 = idx % TT;
    float acc = 0.f;
    for (int n = 0; n < NN; ++n) acc += lhsTN[t * NN + n] * rhs[n * TT + s2];
    prod[idx] = acc;
  }
  __syncthreads();
  for (int idx = tid; idx < TT * TT; idx += blockDim.x) {
    int u = idx / TT, t = idx % TT;
    float acc = 0.f;
    for (int v = 0; v < TT; ++v) {
      float p  = prod[v * TT + t] + b_e[v * TT + t];
      float sg = 1.f / (1.f + expf(-p));
      acc += V_e[u * TT + v] * sg;
    }
    eUT[idx] = acc;
  }
  __syncthreads();
  if (tid < TT) {
    int t = tid;
    float mx = -1e30f;
    for (int u = 0; u < TT; ++u) mx = fmaxf(mx, eUT[u * TT + t]);
    float sm = 0.f;
    for (int u = 0; u < TT; ++u) sm += expf(eUT[u * TT + t] - mx);
    float inv = 1.f / sm;
    for (int u = 0; u < TT; ++u)
      E[(size_t)b * TT * TT + u * TT + t] = expf(eUT[u * TT + t] - mx) * inv;
  }
}

// ---------------- 2. temporal gating: x_TAt = x @ E -------------------------
__global__ void k_gate(const float* __restrict__ x, const float* __restrict__ E,
                       float* __restrict__ xTAt) {
  size_t id = (size_t)blockIdx.x * blockDim.x + threadIdx.x;
  if (id >= (size_t)BB * NN * FF) return;
  int b = (int)(id / (NN * FF));
  const float* xp = x + id * TT;
  const float* Eb = E + (size_t)b * TT * TT;
  float xs[TT];
#pragma unroll
  for (int s = 0; s < TT; ++s) xs[s] = xp[s];
  float* op = xTAt + id * TT;
#pragma unroll
  for (int t = 0; t < TT; ++t) {
    float acc = 0.f;
#pragma unroll
    for (int s = 0; s < TT; ++s) acc += xs[s] * Eb[s * TT + t];
    op[t] = acc;
  }
}

// ---------------- 3. spatial attention prep: lhs2 (B,N,T), rhsS (B,T,N) -----
__global__ void k_sa_prep(const float* __restrict__ xTAt, const float* __restrict__ W1,
                          const float* __restrict__ W2, const float* __restrict__ W3,
                          float* __restrict__ lhs2, float* __restrict__ rhsS) {
  int id = blockIdx.x * blockDim.x + threadIdx.x;
  if (id >= BB * NN) return;
  int b = id / NN, n = id % NN;
  const float* xp = xTAt + (size_t)id * FF * TT;
  float a[FF];
  float rh[TT];
#pragma unroll
  for (int t = 0; t < TT; ++t) rh[t] = 0.f;
  for (int f = 0; f < FF; ++f) {
    float s = 0.f;
    const float* xr = xp + f * TT;
    float w3 = W3[f];
#pragma unroll
    for (int t = 0; t < TT; ++t) { s += xr[t] * W1[t]; rh[t] += w3 * xr[t]; }
    a[f] = s;
  }
  for (int t = 0; t < TT; ++t) {
    float s = 0.f;
    for (int f = 0; f < FF; ++f) s += a[f] * W2[f * TT + t];
    lhs2[(size_t)id * TT + t] = s;
    rhsS[((size_t)b * TT + t) * NN + n] = rh[t];
  }
}

// ---------------- 4. prod + sigmoid -> sig16 (B,NPAD,NPAD f16, padded) ------
__global__ void k_sa_sig(const float* __restrict__ lhs2, const float* __restrict__ rhsS,
                         const float* __restrict__ b_s, _Float16* __restrict__ sig16) {
  size_t id = (size_t)blockIdx.x * blockDim.x + threadIdx.x;
  if (id >= (size_t)BB * NPAD * NPAD) return;
  int v = (int)(id % NPAD);
  int a = (int)((id / NPAD) % NPAD);
  int b = (int)(id / ((size_t)NPAD * NPAD));
  float out = 0.f;
  if (a < NN && v < NN) {
    float p = 0.f;
    const float* lp = lhs2 + ((size_t)b * NN + a) * TT;
    const float* rp = rhsS + (size_t)b * TT * NN + v;
#pragma unroll
    for (int t = 0; t < TT; ++t) p += lp[t] * rp[(size_t)t * NN];
    p += b_s[a * NN + v];
    out = 1.f / (1.f + expf(-p));
  }
  sig16[id] = (_Float16)out;
}

// ---------------- 5. f16 conversions: V_s (Bt form) and Theta^T -------------
__global__ void k_prep_small(const float* __restrict__ V_s, const float* __restrict__ Theta,
                             _Float16* __restrict__ VsP16, _Float16* __restrict__ ThT16) {
  int id = blockIdx.x * blockDim.x + threadIdx.x;
  if (id < NPAD * NPAD) {
    int v = id % NPAD, i = id / NPAD;          // VsP16[i][v] = V_s[i][v] (K=v contiguous)
    float val = (i < NN && v < NN) ? V_s[i * NN + v] : 0.f;
    VsP16[id] = (_Float16)val;
  }
  int id2 = id - NPAD * NPAD;
  if (id2 >= 0 && id2 < KK * CC * FF) {        // ThT16[k][c][f] = Theta[k][f][c]
    int k = id2 / (CC * FF);
    int rem = id2 % (CC * FF);
    int c = rem / FF, f = rem % FF;
    ThT16[id2] = (_Float16)Theta[((size_t)k * FF + f) * CC + c];
  }
}

// ---------------- 5b. f16 conv weights (Bt form, K contiguous) --------------
__global__ void k_wprep(const float* __restrict__ rconv_w, const float* __restrict__ tconv_w,
                        _Float16* __restrict__ rw16, _Float16* __restrict__ tw16) {
  int id = blockIdx.x * blockDim.x + threadIdx.x;
  if (id < CTO * FF) rw16[id] = (_Float16)rconv_w[id];   // rw16[co][f]
  int id2 = id - CTO * FF;
  if (id2 >= 0 && id2 < 3 * CTO * CC) {                  // tw16[tap][co][ci]
    int tap = id2 / (CTO * CC);
    int rem = id2 % (CTO * CC);                          // rem = co*CC + ci
    tw16[id2] = (_Float16)tconv_w[rem * 3 + tap];
  }
}

// ---------------- 6. WMMA: S_pre[b] = sig[b] (NxN) . V_s^T (NxN) ------------
__global__ void __launch_bounds__(256)
k_sgemm_sattn(const _Float16* __restrict__ sig16, const _Float16* __restrict__ VsP16,
              float* __restrict__ S) {
  const int lane = threadIdx.x & 31;
  const int wv   = threadIdx.x >> 5;
  const int tile = blockIdx.x * 8 + wv;
  const int MT = NPAD / 16, NT2 = NPAD / 16;
  int b = tile / (MT * NT2);
  if (b >= BB) return;
  int r  = tile % (MT * NT2);
  int m0 = (r / NT2) * 16, n0 = (r % NT2) * 16;
  const _Float16* A = sig16 + (size_t)b * NPAD * NPAD;
  v8f acc = {};
  for (int k0 = 0; k0 < NPAD; k0 += 32) {
    v16h a  = load_frag(A, NPAD, m0, k0, lane);
    v16h bf = load_frag(VsP16, NPAD, n0, k0, lane);
    acc = __builtin_amdgcn_wmma_f32_16x16x32_f16(false, a, false, bf,
                                                 (short)0, acc, false, false);
  }
  int col   = n0 + (lane & 15);
  int rbase = m0 + ((lane >> 4) << 3);
  float* D = S + (size_t)b * NPAD * NPAD;
#pragma unroll
  for (int v = 0; v < 8; ++v) D[(size_t)(rbase + v) * NPAD + col] = acc[v];
}

// ---------------- 7. softmax over axis a (rows) per (b, i-column) -----------
__global__ void k_softmax_S(float* __restrict__ S) {
  __shared__ float red[64];
  int b = blockIdx.x / NN, icol = blockIdx.x % NN;
  int tid = threadIdx.x;
  float* col = S + (size_t)b * NPAD * NPAD + icol;
  float mx = -1e30f;
  for (int a = tid; a < NN; a += 64) mx = fmaxf(mx, col[(size_t)a * NPAD]);
  red[tid] = mx; __syncthreads();
  for (int s = 32; s > 0; s >>= 1) { if (tid < s) red[tid] = fmaxf(red[tid], red[tid + s]); __syncthreads(); }
  mx = red[0]; __syncthreads();
  float sm = 0.f;
  for (int a = tid; a < NN; a += 64) sm += expf(col[(size_t)a * NPAD] - mx);
  red[tid] = sm; __syncthreads();
  for (int s = 32; s > 0; s >>= 1) { if (tid < s) red[tid] += red[tid + s]; __syncthreads(); }
  float inv = 1.f / red[0];
  for (int a = tid; a < NN; a += 64)
    col[(size_t)a * NPAD] = expf(col[(size_t)a * NPAD] - mx) * inv;
}

// ---------------- 8. x -> f16, layout (B,T,NPAD,F) --------------------------
__global__ void k_xT16(const float* __restrict__ x, _Float16* __restrict__ xT16) {
  size_t id = (size_t)blockIdx.x * blockDim.x + threadIdx.x;
  if (id >= (size_t)BB * TT * NPAD * FF) return;
  int f = (int)(id % FF);
  int i = (int)((id / FF) % NPAD);
  int t = (int)((id / ((size_t)FF * NPAD)) % TT);
  int b = (int)(id / ((size_t)FF * NPAD * TT));
  float val = 0.f;
  if (i < NN) val = x[(((size_t)b * NN + i) * FF + f) * TT + t];
  xT16[id] = (_Float16)val;
}

// ---------------- 9. WMMA: xTh^T[k,b,(c,t),i] = (x_t . Theta_k)^T -----------
__global__ void __launch_bounds__(256)
k_xtheta_gemm(const _Float16* __restrict__ xT16, const _Float16* __restrict__ ThT16,
              _Float16* __restrict__ XThT) {
  const int lane = threadIdx.x & 31;
  const int wv   = threadIdx.x >> 5;
  const int tile = blockIdx.x * 8 + wv;
  const int per = (NPAD / 16) * (CC / 16);   // 80 tiles per (k,b,t)
  int kbt = tile / per;
  if (kbt >= KK * BB * TT) return;
  int r  = tile % per;
  int m0 = (r / (CC / 16)) * 16, n0 = (r % (CC / 16)) * 16;
  int k = kbt / (BB * TT);
  int bt = kbt % (BB * TT);
  int b = bt / TT, t = bt % TT;
  const _Float16* A  = xT16  + (size_t)(b * TT + t) * NPAD * FF;
  const _Float16* Bt = ThT16 + (size_t)k * CC * FF;
  v8f acc = {};
  for (int k0 = 0; k0 < FF; k0 += 32) {
    v16h a  = load_frag(A, FF, m0, k0, lane);
    v16h bf = load_frag(Bt, FF, n0, k0, lane);
    acc = __builtin_amdgcn_wmma_f32_16x16x32_f16(false, a, false, bf,
                                                 (short)0, acc, false, false);
  }
  // transposed store: row = ct = c*T+t (lane column), cols = 8 contiguous i
  int col   = n0 + (lane & 15);               // channel c
  int rbase = m0 + ((lane >> 4) << 3);        // node i
  _Float16* D = XThT + (size_t)(k * BB + b) * CTT * NPAD;
  v8h pk;
#pragma unroll
  for (int v = 0; v < 8; ++v) pk[v] = (_Float16)acc[v];
  *(v8h*)(D + (size_t)(col * TT + t) * NPAD + rbase) = pk;
}

// ---------------- 10. A^T prep: AT16[k,b,j,i] = cheb[k,i,j]*S[b,i,j] --------
__global__ void k_AT_prep(const float* __restrict__ cheb, const float* __restrict__ S,
                          _Float16* __restrict__ AT16) {
  size_t id = (size_t)blockIdx.x * blockDim.x + threadIdx.x;
  if (id >= (size_t)KK * BB * NPAD * NPAD) return;
  int i  = (int)(id % NPAD);
  int j  = (int)((id / NPAD) % NPAD);
  int kb = (int)(id / ((size_t)NPAD * NPAD));
  int k = kb / BB, b = kb % BB;
  float val = 0.f;
  if (i < NN && j < NN)
    val = cheb[((size_t)k * NN + i) * NN + j] * S[((size_t)b * NPAD + i) * NPAD + j];
  AT16[((size_t)kb * NPAD + j) * NPAD + i] = (_Float16)val;
}

// ---------------- 11. main WMMA GEMM: gcn[b] = relu(sum_k A_k^T . xTh_k) ----
__global__ void __launch_bounds__(256)
k_cheb_gemm(const _Float16* __restrict__ AT16, const _Float16* __restrict__ XThT,
            float* __restrict__ gcn) {
  const int lane = threadIdx.x & 31;
  const int wv   = threadIdx.x >> 5;
  const int tile = blockIdx.x * 8 + wv;
  const int per = (NPAD / 16) * (CTT / 16);   // 20*48 = 960 tiles per batch
  int b = tile / per;
  if (b >= BB) return;
  int r  = tile % per;
  int m0 = (r / (CTT / 16)) * 16, n0 = (r % (CTT / 16)) * 16;
  v8f acc = {};
  for (int k = 0; k < KK; ++k) {
    const _Float16* A  = AT16 + (size_t)(k * BB + b) * NPAD * NPAD;
    const _Float16* Bt = XThT + (size_t)(k * BB + b) * CTT * NPAD;
    for (int k0 = 0; k0 < NPAD; k0 += 32) {
      v16h a  = load_frag(A, NPAD, m0, k0, lane);
      v16h bf = load_frag(Bt, NPAD, n0, k0, lane);
      acc = __builtin_amdgcn_wmma_f32_16x16x32_f16(false, a, false, bf,
                                                   (short)0, acc, false, false);
    }
  }
  int col   = n0 + (lane & 15);
  int rbase = m0 + ((lane >> 4) << 3);
#pragma unroll
  for (int v = 0; v < 8; ++v) {
    int j = rbase + v;
    if (j < NN) {
      float val = acc[v];
      gcn[((size_t)b * NN + j) * CTT + col] = val > 0.f ? val : 0.f;
    }
  }
}

// ---------------- 12. relu'd gcn -> f16, layout (B,T,NPAD,C) ----------------
__global__ void k_g16(const float* __restrict__ gcn, _Float16* __restrict__ G16) {
  size_t id = (size_t)blockIdx.x * blockDim.x + threadIdx.x;
  if (id >= (size_t)BB * TT * NPAD * CC) return;
  int ci = (int)(id % CC);
  int i  = (int)((id / CC) % NPAD);
  int t  = (int)((id / ((size_t)CC * NPAD)) % TT);
  int b  = (int)(id / ((size_t)CC * NPAD * TT));
  float v = 0.f;
  if (i < NN) v = gcn[((size_t)b * NN + i) * CTT + ci * TT + t];
  G16[id] = (_Float16)v;
}

// ---------------- 13. WMMA tail: res + (1,3) tconv, bias, ReLU, LayerNorm ---
// One wave per (b, t, 16-node slab); each wave computes all 64 channels
// (4 accumulator tiles), stages the slab in LDS and applies channel-axis LN.
__global__ void __launch_bounds__(256)
k_tail(const _Float16* __restrict__ xT16, const _Float16* __restrict__ G16,
       const _Float16* __restrict__ rw16, const _Float16* __restrict__ tw16,
       const float* __restrict__ tconv_b, const float* __restrict__ rconv_b,
       const float* __restrict__ gamma, const float* __restrict__ beta,
       float* __restrict__ out) {
  __shared__ float hsh[8][16][CTO + 1];   // per-wave 16 rows x 64 channels
  __shared__ float mur[8][16][2];         // per-row mean / rsqrt(var)
  const int lane = threadIdx.x & 31;
  const int wv   = threadIdx.x >> 5;
  const int wid  = blockIdx.x * 8 + wv;   // exact grid: BB*TT*(NPAD/16)
  const int it = wid % (NPAD / 16);
  const int bt = wid / (NPAD / 16);
  const int b = bt / TT, t = bt % TT;
  const int m0 = it * 16;
  const _Float16* Ax = xT16 + (size_t)(b * TT + t) * NPAD * FF;

  v8f acc[4];
#pragma unroll
  for (int g = 0; g < 4; ++g) {
    int n0 = g * 16;
    v8f a = {};
    // 1x1 residual conv: x_t (NPAD x F) . rconv_w^T
    for (int k0 = 0; k0 < FF; k0 += 32) {
      v16h af = load_frag(Ax, FF, m0, k0, lane);
      v16h bf = load_frag(rw16, FF, n0, k0, lane);
      a = __builtin_amdgcn_wmma_f32_16x16x32_f16(false, af, false, bf,
                                                 (short)0, a, false, false);
    }
    // (1,3) temporal conv, SAME padding: sum_tap G_{t+tap-1} . w_tap^T
    for (int tap = 0; tap < 3; ++tap) {
      int tp = t + tap - 1;
      if (tp < 0 || tp >= TT) continue;
      const _Float16* Ag = G16  + (size_t)(b * TT + tp) * NPAD * CC;
      const _Float16* Bw = tw16 + (size_t)tap * CTO * CC;
      for (int k0 = 0; k0 < CC; k0 += 32) {
        v16h af = load_frag(Ag, CC, m0, k0, lane);
        v16h bf = load_frag(Bw, CC, n0, k0, lane);
        a = __builtin_amdgcn_wmma_f32_16x16x32_f16(false, af, false, bf,
                                                   (short)0, a, false, false);
      }
    }
    acc[g] = a;
  }
  // bias + ReLU -> LDS
  const int colL  = lane & 15;
  const int rbase = (lane >> 4) << 3;
#pragma unroll
  for (int g = 0; g < 4; ++g) {
    int co = g * 16 + colL;
    float bias = tconv_b[co] + rconv_b[co];
#pragma unroll
    for (int v = 0; v < 8; ++v) {
      float val = acc[g][v] + bias;
      hsh[wv][rbase + v][co] = val > 0.f ? val : 0.f;
    }
  }
  __syncthreads();
  if (lane < 16) {
    float m = 0.f;
    for (int c = 0; c < CTO; ++c) m += hsh[wv][lane][c];
    m *= (1.f / CTO);
    float var = 0.f;
    for (int c = 0; c < CTO; ++c) { float d = hsh[wv][lane][c] - m; var += d * d; }
    var *= (1.f / CTO);
    mur[wv][lane][0] = m;
    mur[wv][lane][1] = rsqrtf(var + LN_EPS);
  }
  __syncthreads();
  // each lane writes channels {lane, lane+32} for all 16 rows of the slab
#pragma unroll 1
  for (int row = 0; row < 16; ++row) {
    int i = m0 + row;
    if (i >= NN) continue;
    float m = mur[wv][row][0], rsv = mur[wv][row][1];
#pragma unroll
    for (int h = 0; h < 2; ++h) {
      int co = lane + 32 * h;
      float v = hsh[wv][row][co];
      out[(((size_t)b * NN + i) * CTO + co) * TT + t] =
          (v - m) * rsv * gamma[co] + beta[co];
    }
  }
}

// ---------------------------------------------------------------------------
static inline size_t align256(size_t x) { return (x + 255) & ~(size_t)255; }

extern "C" void kernel_launch(void* const* d_in, const int* in_sizes, int n_in,
                              void* d_out, int out_size, void* d_ws, size_t ws_size,
                              hipStream_t stream) {
  (void)in_sizes; (void)n_in; (void)out_size; (void)ws_size;
  const float* x       = (const float*)d_in[0];
  const float* W1      = (const float*)d_in[1];
  const float* W2      = (const float*)d_in[2];
  const float* W3      = (const float*)d_in[3];
  const float* b_s     = (const float*)d_in[4];
  const float* V_s     = (const float*)d_in[5];
  const float* U1      = (const float*)d_in[6];
  const float* U2      = (const float*)d_in[7];
  const float* U3      = (const float*)d_in[8];
  const float* b_e     = (const float*)d_in[9];
  const float* V_e     = (const float*)d_in[10];
  const float* cheb    = (const float*)d_in[11];
  const float* Theta   = (const float*)d_in[12];
  const float* tconv_w = (const float*)d_in[13];
  const float* tconv_b = (const float*)d_in[14];
  const float* rconv_w = (const float*)d_in[15];
  const float* rconv_b = (const float*)d_in[16];
  const float* gamma   = (const float*)d_in[17];
  const float* beta    = (const float*)d_in[18];
  float* out = (float*)d_out;

  char* ws = (char*)d_ws;
  size_t off = 0;
  auto alloc = [&](size_t bytes) -> char* {
    char* p = ws + off;
    off = align256(off + bytes);
    return p;
  };
  float*    E     = (float*)   alloc((size_t)BB * TT * TT * 4);
  float*    xTAt  = (float*)   alloc((size_t)BB * NN * FF * TT * 4);  // aliased as gcn later
  float*    lhs2  = (float*)   alloc((size_t)BB * NN * TT * 4);
  float*    rhsS  = (float*)   alloc((size_t)BB * NN * TT * 4);
  _Float16* sig16 = (_Float16*)alloc((size_t)BB * NPAD * NPAD * 2);   // aliased as G16 later
  _Float16* VsP16 = (_Float16*)alloc((size_t)NPAD * NPAD * 2);
  _Float16* ThT16 = (_Float16*)alloc((size_t)KK * CC * FF * 2);
  float*    S     = (float*)   alloc((size_t)BB * NPAD * NPAD * 4);
  _Float16* xT16  = (_Float16*)alloc((size_t)BB * TT * NPAD * FF * 2);
  _Float16* XThT  = (_Float16*)alloc((size_t)KK * BB * CTT * NPAD * 2);
  _Float16* AT16  = (_Float16*)alloc((size_t)KK * BB * NPAD * NPAD * 2);
  _Float16* rw16  = (_Float16*)alloc((size_t)CTO * FF * 2);
  _Float16* tw16  = (_Float16*)alloc((size_t)3 * CTO * CC * 2);
  // aliases (regions dead by the time these are written):
  float*    gcn = xTAt;                 // B*N*C*T == B*N*F*T
  _Float16* G16 = sig16;                // needs 31.5MB; sig16..S span is 39.5MB

  // 1. temporal attention
  k_temporal_attn<<<BB, 256, 0, stream>>>(x, U1, U2, U3, b_e, V_e, E);
  // 2. temporal gating
  {
    size_t n = (size_t)BB * NN * FF;
    k_gate<<<(int)((n + 255) / 256), 256, 0, stream>>>(x, E, xTAt);
  }
  // 3-4. spatial attention prod + sigmoid (f16, padded)
  k_sa_prep<<<(BB * NN + 255) / 256, 256, 0, stream>>>(xTAt, W1, W2, W3, lhs2, rhsS);
  {
    size_t n = (size_t)BB * NPAD * NPAD;
    k_sa_sig<<<(int)((n + 255) / 256), 256, 0, stream>>>(lhs2, rhsS, b_s, sig16);
  }
  // 5. f16 operand prep
  k_prep_small<<<(NPAD * NPAD + KK * CC * FF + 255) / 256, 256, 0, stream>>>(V_s, Theta, VsP16, ThT16);
  k_wprep<<<(CTO * FF + 3 * CTO * CC + 255) / 256, 256, 0, stream>>>(rconv_w, tconv_w, rw16, tw16);
  // 6. WMMA: S_pre = sig . V_s^T  (per batch)
  k_sgemm_sattn<<<(BB * (NPAD / 16) * (NPAD / 16)) / 8, 256, 0, stream>>>(sig16, VsP16, S);
  // 7. softmax over node axis
  k_softmax_S<<<BB * NN, 64, 0, stream>>>(S);
  // 8. x -> f16 (B,T,NPAD,F)
  {
    size_t n = (size_t)BB * TT * NPAD * FF;
    k_xT16<<<(int)((n + 255) / 256), 256, 0, stream>>>(x, xT16);
  }
  // 9. WMMA: xTheta (stored transposed, per k,b,t)
  k_xtheta_gemm<<<(KK * BB * TT * (NPAD / 16) * (CC / 16)) / 8, 256, 0, stream>>>(xT16, ThT16, XThT);
  // 10. A^T = cheb * S (f16, transposed)
  {
    size_t n = (size_t)KK * BB * NPAD * NPAD;
    k_AT_prep<<<(int)((n + 255) / 256), 256, 0, stream>>>(cheb, S, AT16);
  }
  // 11. WMMA main GEMM + ReLU -> gcn (aliases xTAt, dead by now)
  k_cheb_gemm<<<(BB * (NPAD / 16) * (CTT / 16)) / 8, 256, 0, stream>>>(AT16, XThT, gcn);
  // 12. relu'd gcn -> f16 (aliases sig16/S span, dead by now)
  {
    size_t n = (size_t)BB * TT * NPAD * CC;
    k_g16<<<(int)((n + 255) / 256), 256, 0, stream>>>(gcn, G16);
  }
  // 13. WMMA conv tail + LayerNorm -> out
  k_tail<<<(BB * TT * (NPAD / 16)) / 8, 256, 0, stream>>>(xT16, G16, rw16, tw16,
                                                          tconv_b, rconv_b, gamma, beta, out);
}
